// MSAColGlobalAttention_58548994179790
// MI455X (gfx1250) — matrix operand order repaired and checked by reading, not compiled, not run on hardware
//
#include <hip/hip_runtime.h>
#include <hip/hip_bf16.h>

typedef __attribute__((ext_vector_type(16))) _Float16 v16h;
typedef __attribute__((ext_vector_type(8)))  _Float16 v8h;
typedef __attribute__((ext_vector_type(8)))  float    v8f;

#define Nn 512
#define Ll 1024
#define Dd 64
#define NSPLIT 32
#define NCHUNK (Nn / NSPLIT)   // 16
#define SCALE_F 0.3535533906f  // 1/sqrt(8)

// ---- WMMA helpers -----------------------------------------------------------

__device__ __forceinline__ v8f wmma16(v16h a, v16h b, v8f c) {
    return __builtin_amdgcn_wmma_f32_16x16x32_f16(
        /*neg_a=*/false, a, /*neg_b=*/false, b,
        /*c_mod=*/(short)0, c, /*reuse_a=*/false, /*reuse_b=*/false);
}

// A fragment: rows m = lane&15 of a row-major 16x64 f16 tile, K-chunk c0 (0/32).
// element e -> K = c0 + (e<8 ? kg*8+e : 16+kg*8+(e-8)), kg = lane>>4.
__device__ __forceinline__ v16h load_a_frag(const _Float16* x16, int lane, int c0) {
    int m = lane & 15, kg = lane >> 4;
    const _Float16* row = x16 + m * 64;
    v8h lo = *(const v8h*)(row + c0 + kg * 8);
    v8h hi = *(const v8h*)(row + c0 + 16 + kg * 8);
    v16h a;
#pragma unroll
    for (int i = 0; i < 8; ++i) { a[i] = lo[i]; a[i + 8] = hi[i]; }
    return a;
}

// B fragment from torch-style weight W[out][64] (f32), output cols e0..e0+15,
// K-chunk c0. element e -> K = c0 + (e<8 ? kg*8+e : 16+kg*8+(e-8)), kg = lane>>4.
__device__ __forceinline__ v16h load_b_frag(const float* __restrict__ W,
                                            int lane, int e0, int c0) {
    int nn = lane & 15, kg = lane >> 4;
    const float* p = W + (size_t)(e0 + nn) * 64 + c0 + kg * 8;
    v16h b;
#pragma unroll
    for (int i = 0; i < 8; ++i) { b[i] = (_Float16)p[i]; b[i + 8] = (_Float16)p[16 + i]; }
    return b;
}

// Packed k|v B fragment: cols 0..7 from Wk[8][64], cols 8..15 from Wv[8][64].
__device__ __forceinline__ v16h load_b_frag_kv(const float* __restrict__ Wk,
                                               const float* __restrict__ Wv,
                                               int lane, int c0) {
    int nn = lane & 15, kg = lane >> 4;
    const float* base = (nn < 8) ? (Wk + (size_t)nn * 64)
                                 : (Wv + (size_t)(nn - 8) * 64);
    const float* p = base + c0 + kg * 8;
    v16h b;
#pragma unroll
    for (int i = 0; i < 8; ++i) { b[i] = (_Float16)p[i]; b[i + 8] = (_Float16)p[16 + i]; }
    return b;
}

// Per-lane LN of a half-row held in registers; writes the f16 half-row to LDS.
// Each lane owns row m = lane&15, channels kg*32..kg*32+31.
__device__ __forceinline__ void ln_halfrow_to_lds(
    const float* __restrict__ src_halfrow,   // global, 32 contiguous f32
    const float* lw, const float* lb,        // per-lane 32 ln params
    _Float16* x16, int m, int kg) {
    float xr[32];
#pragma unroll
    for (int j = 0; j < 8; ++j)
        *(float4*)&xr[j * 4] = *(const float4*)&src_halfrow[j * 4];
    float s = 0.f;
#pragma unroll
    for (int c = 0; c < 32; ++c) s += xr[c];
    s += __shfl_xor(s, 16, 32);                      // combine row halves
    const float mu = s * (1.0f / 64.0f);
    float vv = 0.f;
#pragma unroll
    for (int c = 0; c < 32; ++c) { float d = xr[c] - mu; vv += d * d; }
    vv += __shfl_xor(vv, 16, 32);
    const float inv = rsqrtf(vv * (1.0f / 64.0f) + 1e-5f);
    _Float16* drow = x16 + m * 64 + kg * 32;
#pragma unroll
    for (int j = 0; j < 4; ++j) {
        v8h h;
#pragma unroll
        for (int c = 0; c < 8; ++c)
            h[c] = (_Float16)(((xr[j * 8 + c] - mu) * inv) * lw[j * 8 + c] + lb[j * 8 + c]);
        *(v8h*)(drow + j * 8) = h;
    }
}

// ---- kernel 0: zero qsum ----------------------------------------------------

__global__ void k0_zero(float* __restrict__ p, int count) {
    int i = blockIdx.x * blockDim.x + threadIdx.x;
    if (i < count) p[i] = 0.0f;
}

// ---- kernel 1: LN + q/k/v projections (WMMA), q accumulated over n ----------

__global__ __launch_bounds__(32)
void k1_ln_qkv(const float* __restrict__ msa,
               const float* __restrict__ ln_w, const float* __restrict__ ln_b,
               const float* __restrict__ Wq, const float* __restrict__ Wk,
               const float* __restrict__ Wv,
               float* __restrict__ kT, float* __restrict__ vT,
               float* __restrict__ qsum) {
    __shared__ __align__(16) _Float16 x16[16 * 64];

    const int lane = threadIdx.x;
    const int l0 = blockIdx.x * 16;
    const int n0 = blockIdx.y * NCHUNK;
    const int m = lane & 15, kg = lane >> 4;
    const int nn = m;  // B/C column index alias

    // per-lane LN params for this half-row
    float lw[32], lb[32];
    {
        const float* pw = ln_w + kg * 32;
        const float* pb = ln_b + kg * 32;
#pragma unroll
        for (int j = 0; j < 8; ++j) {
            *(float4*)&lw[j * 4] = *(const float4*)&pw[j * 4];
            *(float4*)&lb[j * 4] = *(const float4*)&pb[j * 4];
        }
    }

    // weight fragments (resident in registers for the whole n-loop)
    v16h bq[4][2];
#pragma unroll
    for (int t = 0; t < 4; ++t)
#pragma unroll
        for (int c = 0; c < 2; ++c) bq[t][c] = load_b_frag(Wq, lane, t * 16, c * 32);
    v16h bkv[2];
#pragma unroll
    for (int c = 0; c < 2; ++c) bkv[c] = load_b_frag_kv(Wk, Wv, lane, c * 32);

    v8f qacc[4];
#pragma unroll
    for (int t = 0; t < 4; ++t) qacc[t] = (v8f){};

    // this lane's half-row within the (n, l-tile) block
    const float* src = msa + (((size_t)n0 * Ll + l0) + m) * 64 + kg * 32;

    for (int ni = 0; ni < NCHUNK; ++ni) {
        if (ni + 1 < NCHUNK)
            __builtin_prefetch(src + (size_t)Ll * 64, 0, 0);  // next n tile

        ln_halfrow_to_lds(src, lw, lb, x16, m, kg);
        __syncthreads();

        v16h a0 = load_a_frag(x16, lane, 0);
        v16h a1 = load_a_frag(x16, lane, 32);

#pragma unroll
        for (int t = 0; t < 4; ++t) {
            qacc[t] = wmma16(a0, bq[t][0], qacc[t]);
            qacc[t] = wmma16(a1, bq[t][1], qacc[t]);
        }
        v8f kv = (v8f){};
        kv = wmma16(a0, bkv[0], kv);
        kv = wmma16(a1, bkv[1], kv);

        // store k/v transposed: [l][n][d], d contiguous
        {
            const int n = n0 + ni;
            const int d = nn & 7;
            float* dstb = (nn < 8) ? kT : vT;
#pragma unroll
            for (int r = 0; r < 8; ++r) {
                const int l = l0 + kg * 8 + r;
                dstb[((size_t)l * Nn + n) * 8 + d] = kv[r];
            }
        }
        src += (size_t)Ll * 64;
        __syncthreads();   // x16 reused next iteration
    }

    // flush q partial sums (NSPLIT partials per (l,e))
#pragma unroll
    for (int t = 0; t < 4; ++t)
#pragma unroll
        for (int r = 0; r < 8; ++r) {
            const int l = l0 + kg * 8 + r;
            const int e = t * 16 + nn;
            unsafeAtomicAdd(&qsum[(size_t)l * 64 + e], qacc[t][r]);
        }
}

// ---- kernel 2: softmax attention over N, per (l, head) ----------------------

__global__ __launch_bounds__(256)
void k2_attn(const float* __restrict__ kT, const float* __restrict__ vT,
             const float* __restrict__ qsum, float* __restrict__ outcol) {
    const int l = blockIdx.x;
    const int h = threadIdx.x >> 5;   // 8 waves = 8 heads
    const int lane = threadIdx.x & 31;

    const float scl = SCALE_F / (float)Nn;   // mean over N then * SCALE
    float qh[8];
#pragma unroll
    for (int d = 0; d < 8; ++d) qh[d] = qsum[(size_t)l * 64 + h * 8 + d] * scl;

    const float* kb = kT + (size_t)l * Nn * 8;
    const float* vb = vT + (size_t)l * Nn * 8;

    float sv[16];
    float mx = -3.0e38f;
#pragma unroll
    for (int j = 0; j < 16; ++j) {
        const size_t n = lane + 32 * j;
        float4 ka = *(const float4*)&kb[n * 8];
        float4 kc = *(const float4*)&kb[n * 8 + 4];
        float s = qh[0] * ka.x + qh[1] * ka.y + qh[2] * ka.z + qh[3] * ka.w
                + qh[4] * kc.x + qh[5] * kc.y + qh[6] * kc.z + qh[7] * kc.w;
        sv[j] = s;
        mx = fmaxf(mx, s);
    }
#pragma unroll
    for (int off = 16; off >= 1; off >>= 1)
        mx = fmaxf(mx, __shfl_xor(mx, off, 32));

    float sum = 0.f, oacc[8];
#pragma unroll
    for (int d = 0; d < 8; ++d) oacc[d] = 0.f;
#pragma unroll
    for (int j = 0; j < 16; ++j) {
        const size_t n = lane + 32 * j;
        const float p = __expf(sv[j] - mx);
        sum += p;
        float4 va = *(const float4*)&vb[n * 8];
        float4 vc = *(const float4*)&vb[n * 8 + 4];
        oacc[0] += p * va.x; oacc[1] += p * va.y;
        oacc[2] += p * va.z; oacc[3] += p * va.w;
        oacc[4] += p * vc.x; oacc[5] += p * vc.y;
        oacc[6] += p * vc.z; oacc[7] += p * vc.w;
    }
#pragma unroll
    for (int off = 16; off >= 1; off >>= 1) {
        sum += __shfl_xor(sum, off, 32);
#pragma unroll
        for (int d = 0; d < 8; ++d) oacc[d] += __shfl_xor(oacc[d], off, 32);
    }
    if (lane == 0) {
        const float inv = 1.0f / sum;
#pragma unroll
        for (int d = 0; d < 8; ++d)
            outcol[(size_t)l * 64 + h * 8 + d] = oacc[d] * inv;
    }
}

// ---- kernel 3: LN + gate (WMMA) * outcol, then out-proj (WMMA) + bo ---------

__global__ __launch_bounds__(32)
void k3_gate_out(const float* __restrict__ msa,
                 const float* __restrict__ ln_w, const float* __restrict__ ln_b,
                 const float* __restrict__ Wg, const float* __restrict__ bg,
                 const float* __restrict__ Wo, const float* __restrict__ bo,
                 const float* __restrict__ outcol, float* __restrict__ out) {
    __shared__ __align__(16) _Float16 x16[16 * 64];
    __shared__ __align__(16) _Float16 t16[16 * 64];
    __shared__ __align__(16) float    oc[16 * 64];

    const int lane = threadIdx.x;
    const int l0 = blockIdx.x * 16;
    const int n0 = blockIdx.y * NCHUNK;
    const int m = lane & 15, kg = lane >> 4;
    const int nn = m;

    // broadcast attention-output tile for these 16 l's (same for all n)
#pragma unroll
    for (int j = 0; j < 8; ++j) {
        const int i = lane + 32 * j;
        *(float4*)&oc[i * 4] = *(const float4*)&outcol[(size_t)l0 * 64 + i * 4];
    }

    float lw[32], lb[32];
    {
        const float* pw = ln_w + kg * 32;
        const float* pb = ln_b + kg * 32;
#pragma unroll
        for (int j = 0; j < 8; ++j) {
            *(float4*)&lw[j * 4] = *(const float4*)&pw[j * 4];
            *(float4*)&lb[j * 4] = *(const float4*)&pb[j * 4];
        }
    }

    v16h bgf[4][2], bof[4][2];
#pragma unroll
    for (int t = 0; t < 4; ++t)
#pragma unroll
        for (int c = 0; c < 2; ++c) {
            bgf[t][c] = load_b_frag(Wg, lane, t * 16, c * 32);
            bof[t][c] = load_b_frag(Wo, lane, t * 16, c * 32);
        }
    float bgv[4], bov[4];
#pragma unroll
    for (int t = 0; t < 4; ++t) {
        bgv[t] = bg[t * 16 + nn];
        bov[t] = bo[t * 16 + nn];
    }
    __syncthreads();

    const float* src = msa + (((size_t)n0 * Ll + l0) + m) * 64 + kg * 32;

    for (int ni = 0; ni < NCHUNK; ++ni) {
        if (ni + 1 < NCHUNK)
            __builtin_prefetch(src + (size_t)Ll * 64, 0, 0);

        ln_halfrow_to_lds(src, lw, lb, x16, m, kg);
        __syncthreads();

        v16h a0 = load_a_frag(x16, lane, 0);
        v16h a1 = load_a_frag(x16, lane, 32);

        // gate = sigmoid(x Wg^T + bg); t = gate * outcol  -> t16 (f16, row-major)
#pragma unroll
        for (int t = 0; t < 4; ++t) {
            v8f g = (v8f){};
            g = wmma16(a0, bgf[t][0], g);
            g = wmma16(a1, bgf[t][1], g);
#pragma unroll
            for (int r = 0; r < 8; ++r) {
                const int mm = kg * 8 + r;
                const int e = t * 16 + nn;
                const float gv = 1.0f / (1.0f + __expf(-(g[r] + bgv[t])));
                t16[mm * 64 + e] = (_Float16)(gv * oc[mm * 64 + e]);
            }
        }
        __syncthreads();

        v16h ta0 = load_a_frag(t16, lane, 0);
        v16h ta1 = load_a_frag(t16, lane, 32);

        const int n = n0 + ni;
        float* dst = out + ((size_t)n * Ll + l0) * 64;
#pragma unroll
        for (int t = 0; t < 4; ++t) {
            v8f y = (v8f){};
            y = wmma16(ta0, bof[t][0], y);
            y = wmma16(ta1, bof[t][1], y);
#pragma unroll
            for (int r = 0; r < 8; ++r) {
                const int mm = kg * 8 + r;
                dst[(size_t)mm * 64 + t * 16 + nn] = y[r] + bov[t];
            }
        }
        src += (size_t)Ll * 64;
        __syncthreads();
    }
}

// ---- launch -----------------------------------------------------------------

extern "C" void kernel_launch(void* const* d_in, const int* in_sizes, int n_in,
                              void* d_out, int out_size, void* d_ws, size_t ws_size,
                              hipStream_t stream) {
    const float* msa  = (const float*)d_in[0];
    const float* ln_w = (const float*)d_in[1];
    const float* ln_b = (const float*)d_in[2];
    const float* Wq   = (const float*)d_in[3];
    const float* Wk   = (const float*)d_in[4];
    const float* Wv   = (const float*)d_in[5];
    const float* Wg   = (const float*)d_in[6];
    const float* bg   = (const float*)d_in[7];
    const float* Wo   = (const float*)d_in[8];
    const float* bo   = (const float*)d_in[9];
    float* out = (float*)d_out;

    char* ws = (char*)d_ws;
    float* kT     = (float*)(ws);                                   // 16 MiB
    float* vT     = (float*)(ws + (size_t)16 * 1024 * 1024);        // 16 MiB
    float* qsum   = (float*)(ws + (size_t)32 * 1024 * 1024);        // 256 KiB
    float* outcol = (float*)(ws + (size_t)32 * 1024 * 1024 + 256 * 1024);

    // qsum is accumulated with atomics -> must be zeroed every call
    k0_zero<<<(Ll * 64 + 255) / 256, 256, 0, stream>>>(qsum, Ll * 64);

    k1_ln_qkv<<<dim3(Ll / 16, NSPLIT), 32, 0, stream>>>(
        msa, ln_w, ln_b, Wq, Wk, Wv, kT, vT, qsum);

    k2_attn<<<Ll, 256, 0, stream>>>(kT, vT, qsum, outcol);

    k3_gate_out<<<dim3(Ll / 16, NSPLIT), 32, 0, stream>>>(
        msa, ln_w, ln_b, Wg, bg, Wo, bo, outcol, out);
}
